// MultiHeadGATLayer_13005160972435
// MI455X (gfx1250) — matrix operand order repaired and checked by reading, not compiled, not run on hardware
//
#include <hip/hip_runtime.h>

#define N_NODES 100000
#define N_EDGES 1600000
#define F_IN    128
#define N_HEADS 4
#define F_HID   32
#define F_OUT   128   // N_HEADS * F_HID
#define NEG_SLOPE 0.2f

typedef __attribute__((ext_vector_type(16))) _Float16 v16h;
typedef __attribute__((ext_vector_type(8)))  _Float16 v8h;
typedef __attribute__((ext_vector_type(8)))  float    v8f;

// ---------------------------------------------------------------------------
// Order-preserving float<->uint key for unsigned atomicMax-based segment max.
// ---------------------------------------------------------------------------
__device__ __forceinline__ unsigned f2key(float f) {
  unsigned u = __float_as_uint(f);
  return (u & 0x80000000u) ? ~u : (u | 0x80000000u);
}
__device__ __forceinline__ float key2f(unsigned k) {
  unsigned u = (k & 0x80000000u) ? (k & 0x7FFFFFFFu) : ~k;
  return __uint_as_float(u);
}

// ---------------------------------------------------------------------------
// K0: Wt[n][k] = (f16) W[k][n]  -- 32KB, done once; makes every B fragment a
// pair of 16-byte runs (2x global_load_b128) instead of 16 strided b32s.
// ---------------------------------------------------------------------------
__global__ __launch_bounds__(256) void gat_w_to_f16t(
    const float* __restrict__ W, _Float16* __restrict__ Wt)
{
  const int i = blockIdx.x * 256 + threadIdx.x;   // i = n*128 + k
  if (i >= F_IN * F_OUT) return;
  const int n = i >> 7, k = i & 127;
  Wt[i] = (_Float16)W[k * F_OUT + n];
}

// ---------------------------------------------------------------------------
// K1: h = X @ W + b via v_wmma_f32_16x16x32_f16.
// One wave computes a 16x128 strip of h (8 accumulators of 16x16 f32).
// A (16x32 f16): lane<16 -> row=lane, halves a[0..7]=K kb..kb+7,
//                a[8..15]=K kb+16..kb+23, kb=(lane>=16)*8  -> two 32B runs.
// B (32x16 f16): col=lane&15, same K order as A: kb..kb+7, kb+16..kb+23.
// C/D: N = lane&15, M = j + (lane>=16)*8 for VGPR j.
// ---------------------------------------------------------------------------
__global__ __launch_bounds__(256) void gat_gemm_wmma(
    const float* __restrict__ X, const _Float16* __restrict__ Wt,
    const float* __restrict__ bias, float* __restrict__ H)
{
  const int lane = threadIdx.x & 31;
  const int tile = blockIdx.x * 8 + (threadIdx.x >> 5);
  if (tile >= N_NODES / 16) return;

  const int rowA   = tile * 16 + (lane & 15);
  const int kbaseA = (lane >> 4) * 8;
  const int col    = lane & 15;
  const int kbaseB = (lane >> 4) * 8;

  v8f acc[8] = {};

  #pragma unroll
  for (int kt = 0; kt < F_IN; kt += 32) {
    // ---- A fragment: two contiguous 32B runs -> 4x b128 loads ----
    const float4* xp = (const float4*)(X + rowA * F_IN + kt + kbaseA);
    const float4* xq = (const float4*)(X + rowA * F_IN + kt + 16 + kbaseA);
    const float4 r0 = xp[0], r1 = xp[1], r2 = xq[0], r3 = xq[1];
    v16h a;
    a[0]  = (_Float16)r0.x; a[1]  = (_Float16)r0.y;
    a[2]  = (_Float16)r0.z; a[3]  = (_Float16)r0.w;
    a[4]  = (_Float16)r1.x; a[5]  = (_Float16)r1.y;
    a[6]  = (_Float16)r1.z; a[7]  = (_Float16)r1.w;
    a[8]  = (_Float16)r2.x; a[9]  = (_Float16)r2.y;
    a[10] = (_Float16)r2.z; a[11] = (_Float16)r2.w;
    a[12] = (_Float16)r3.x; a[13] = (_Float16)r3.y;
    a[14] = (_Float16)r3.z; a[15] = (_Float16)r3.w;

    #pragma unroll
    for (int n = 0; n < 8; ++n) {
      // ---- B fragment: two 8-f16 runs of Wt row (K kb.., kb+16..) -> 2x b128 loads ----
      const v8h* wp = (const v8h*)(Wt + (n * 16 + col) * F_IN + kt + kbaseB);
      v16h bf;
      ((v8h*)&bf)[0] = wp[0];
      ((v8h*)&bf)[1] = wp[2];
      acc[n] = __builtin_amdgcn_wmma_f32_16x16x32_f16(
          /*neg_a=*/false, a, /*neg_b=*/false, bf,
          /*c_mod=*/(short)0, acc[n], /*reuse_a=*/false, /*reuse_b=*/false);
    }
  }

  #pragma unroll
  for (int n = 0; n < 8; ++n) {
    const float bv = bias[n * 16 + col];
    #pragma unroll
    for (int j = 0; j < 8; ++j) {
      const int M = j + (lane >> 4) * 8;
      H[(tile * 16 + M) * F_OUT + n * 16 + col] = acc[n][j] + bv;
    }
  }
}

// ---------------------------------------------------------------------------
// K2: per-node attention scores s_i, s_j; residual init out = h;
// init segment-max keys and denominators. One wave per node.
// ---------------------------------------------------------------------------
__global__ __launch_bounds__(256) void gat_scores(
    const float* __restrict__ H, const float* __restrict__ attn,
    float* __restrict__ Si, float* __restrict__ Sj,
    unsigned* __restrict__ Mkey, float* __restrict__ Denom,
    float* __restrict__ Out)
{
  const int lane = threadIdx.x & 31;
  const int node = blockIdx.x * 8 + (threadIdx.x >> 5);
  if (node >= N_NODES) return;

  const int head = lane >> 3;           // 8 lanes per head (8*4 = 32 feats)
  const int f0   = (lane & 7) * 4;

  const float4 h4 = *(const float4*)(H + node * F_OUT + lane * 4);
  *(float4*)(Out + node * F_OUT + lane * 4) = h4;   // residual: out = h

  const float4 ai = *(const float4*)(attn + head * 2 * F_HID + f0);
  const float4 aj = *(const float4*)(attn + head * 2 * F_HID + F_HID + f0);

  float pi = h4.x * ai.x + h4.y * ai.y + h4.z * ai.z + h4.w * ai.w;
  float pj = h4.x * aj.x + h4.y * aj.y + h4.z * aj.z + h4.w * aj.w;
  #pragma unroll
  for (int d = 4; d > 0; d >>= 1) {
    pi += __shfl_down(pi, d, 8);
    pj += __shfl_down(pj, d, 8);
  }
  if ((lane & 7) == 0) {
    Si[node * N_HEADS + head] = pi;
    Sj[node * N_HEADS + head] = pj;
  }
  if (lane < N_HEADS) {
    Mkey[node * N_HEADS + lane]  = 0u;    // key 0 < any real logit key
    Denom[node * N_HEADS + lane] = 0.0f;
  }
}

// ---------------------------------------------------------------------------
// K3: per-edge logits eij = leaky_relu(s_i[tgt] + s_j[src]); segment max
// via unsigned atomicMax on order-preserving keys.
// ---------------------------------------------------------------------------
__global__ __launch_bounds__(256) void gat_edge_logits(
    const long long* __restrict__ EI,
    const float* __restrict__ Si, const float* __restrict__ Sj,
    float* __restrict__ Eij, unsigned* __restrict__ Mkey)
{
  const int e = blockIdx.x * blockDim.x + threadIdx.x;
  if (e >= N_EDGES) return;
  const int tgt = (int)EI[e];
  const int src = (int)EI[N_EDGES + e];

  const float4 si = *(const float4*)(Si + tgt * N_HEADS);
  const float4 sj = *(const float4*)(Sj + src * N_HEADS);
  float v[4] = { si.x + sj.x, si.y + sj.y, si.z + sj.z, si.w + sj.w };
  #pragma unroll
  for (int h = 0; h < N_HEADS; ++h)
    v[h] = (v[h] > 0.0f) ? v[h] : NEG_SLOPE * v[h];

  *(float4*)(Eij + (size_t)e * N_HEADS) = make_float4(v[0], v[1], v[2], v[3]);
  #pragma unroll
  for (int h = 0; h < N_HEADS; ++h)
    atomicMax(&Mkey[tgt * N_HEADS + h], f2key(v[h]));
}

// ---------------------------------------------------------------------------
// K4: ex = exp(eij - m[tgt]); denom[tgt] += ex (stored over Eij buffer).
// ---------------------------------------------------------------------------
__global__ __launch_bounds__(256) void gat_edge_exp(
    const long long* __restrict__ EI,
    float* __restrict__ Eij, const unsigned* __restrict__ Mkey,
    float* __restrict__ Denom)
{
  const int e = blockIdx.x * blockDim.x + threadIdx.x;
  if (e >= N_EDGES) return;
  const int tgt = (int)EI[e];

  float4 l = *(const float4*)(Eij + (size_t)e * N_HEADS);
  float ex[4];
  const float* lp = &l.x;
  #pragma unroll
  for (int h = 0; h < N_HEADS; ++h) {
    const float m = key2f(Mkey[tgt * N_HEADS + h]);
    ex[h] = __expf(lp[h] - m);
  }
  *(float4*)(Eij + (size_t)e * N_HEADS) = make_float4(ex[0], ex[1], ex[2], ex[3]);
  #pragma unroll
  for (int h = 0; h < N_HEADS; ++h)
    atomicAdd(&Denom[tgt * N_HEADS + h], ex[h]);
}

// ---------------------------------------------------------------------------
// K5: out[tgt] += alpha * h[src]. One wave per edge; lane covers 4 features.
// out fits in the 192MB L2, so the f32 atomics resolve on-chip.
// ---------------------------------------------------------------------------
__global__ __launch_bounds__(256) void gat_edge_scatter(
    const long long* __restrict__ EI, const float* __restrict__ H,
    const float* __restrict__ Ex, const float* __restrict__ Denom,
    float* __restrict__ Out)
{
  const int lane = threadIdx.x & 31;
  const int e = blockIdx.x * 8 + (threadIdx.x >> 5);
  if (e >= N_EDGES) return;
  const int tgt  = (int)EI[e];
  const int src  = (int)EI[N_EDGES + e];
  const int head = lane >> 3;

  const float alpha = Ex[(size_t)e * N_HEADS + head] / Denom[tgt * N_HEADS + head];
  const float4 hs = *(const float4*)(H + src * F_OUT + lane * 4);

  float* o = Out + tgt * F_OUT + lane * 4;
  atomicAdd(o + 0, alpha * hs.x);
  atomicAdd(o + 1, alpha * hs.y);
  atomicAdd(o + 2, alpha * hs.z);
  atomicAdd(o + 3, alpha * hs.w);
}

// ---------------------------------------------------------------------------
// Launcher
// ---------------------------------------------------------------------------
extern "C" void kernel_launch(void* const* d_in, const int* in_sizes, int n_in,
                              void* d_out, int out_size, void* d_ws, size_t ws_size,
                              hipStream_t stream) {
  const float*     X    = (const float*)d_in[0];
  const long long* EI   = (const long long*)d_in[1];
  const float*     W    = (const float*)d_in[2];
  const float*     bias = (const float*)d_in[3];
  const float*     attn = (const float*)d_in[4];
  float* Out = (float*)d_out;

  char* ws = (char*)d_ws;
  const size_t SZ_H = (size_t)N_NODES * F_OUT   * sizeof(float);   // 51.2 MB
  const size_t SZ_S = (size_t)N_NODES * N_HEADS * sizeof(float);   // 1.6 MB
  const size_t SZ_E = (size_t)N_EDGES * N_HEADS * sizeof(float);   // 25.6 MB
  float*    H     = (float*)(ws);
  float*    Si    = (float*)(ws + SZ_H);
  float*    Sj    = (float*)(ws + SZ_H + SZ_S);
  unsigned* Mkey  = (unsigned*)(ws + SZ_H + 2 * SZ_S);
  float*    Denom = (float*)(ws + SZ_H + 3 * SZ_S);
  float*    Eij   = (float*)(ws + SZ_H + 4 * SZ_S);
  _Float16* Wt    = (_Float16*)(ws + SZ_H + 4 * SZ_S + SZ_E);      // 32 KB

  const int n_tiles = N_NODES / 16;                       // 6250 (exact)
  gat_w_to_f16t<<<(F_IN * F_OUT + 255) / 256, 256, 0, stream>>>(W, Wt);
  gat_gemm_wmma<<<(n_tiles + 7) / 8, 256, 0, stream>>>(X, Wt, bias, H);
  gat_scores<<<(N_NODES + 7) / 8, 256, 0, stream>>>(H, attn, Si, Sj, Mkey, Denom, Out);
  gat_edge_logits<<<(N_EDGES + 255) / 256, 256, 0, stream>>>(EI, Si, Sj, Eij, Mkey);
  gat_edge_exp<<<(N_EDGES + 255) / 256, 256, 0, stream>>>(EI, Eij, Mkey, Denom);
  gat_edge_scatter<<<(N_EDGES + 7) / 8, 256, 0, stream>>>(EI, H, Eij, Denom, Out);
}
